// LSTMLayerBidirectional_28656021799667
// MI455X (gfx1250) — matrix operand order repaired and checked by reading, not compiled, not run on hardware
//
#include <hip/hip_runtime.h>
#include <hip/hip_bf16.h>

typedef __bf16 bf16;
typedef __attribute__((ext_vector_type(16))) __bf16 v16bf;
typedef __attribute__((ext_vector_type(8)))  __bf16 v8bf;
typedef __attribute__((ext_vector_type(8)))  float  v8f;
typedef __attribute__((ext_vector_type(4)))  unsigned int u32x4;
typedef __attribute__((ext_vector_type(8)))  unsigned int u32x8;

#define T_LEN 512
#define BATCH 32
#define DIN   512
#define HID   512
#define G4    2048   // 4*H (gate-interleaved: n' = 4*h + gate)
#define NDIR  2

__device__ __forceinline__ float sigmoidf_(float x){ return 1.0f/(1.0f + __expf(-x)); }

// Build a v16bf WMMA fragment from two 16B-aligned 8-element chunks.
__device__ __forceinline__ v16bf make_frag(const bf16* lo, const bf16* hi){
  v8bf a = *(const v8bf*)lo;
  v8bf b = *(const v8bf*)hi;
  v16bf r;
  #pragma unroll
  for (int i=0;i<8;i++){ r[i]=a[i]; r[i+8]=b[i]; }
  return r;
}

// Per-lane async 16B copy global -> LDS (ASYNCcnt path, GVS addressing).
__device__ __forceinline__ void async_load_b128(unsigned int lds_addr, unsigned int goff,
                                                unsigned long long sbase){
  asm volatile("global_load_async_to_lds_b128 %0, %1, %2"
               :: "v"(lds_addr), "v"(goff), "s"(sbase)
               : "memory");
}

__device__ __forceinline__ void wait_async_all(){
  asm volatile("s_wait_asynccnt 0x0" ::: "memory");
}

// ---------------- setup kernels ----------------

__global__ void k_cvt_x(const float* __restrict__ x, bf16* __restrict__ xb){
  int i = (blockIdx.x*blockDim.x + threadIdx.x)*4;   // n = T*B*D, divisible by 1024
  float4 v = *(const float4*)(x + i);
  xb[i+0] = (bf16)v.x; xb[i+1] = (bf16)v.y; xb[i+2] = (bf16)v.z; xb[i+3] = (bf16)v.w;
}

// Permute gate rows (n = g*H+h  ->  n' = 4*h+g) for both weight matrices + biases.
__global__ void k_cvt_w(const float* __restrict__ Wih, const float* __restrict__ bih,
                        const float* __restrict__ Whh, const float* __restrict__ bhh,
                        bf16* __restrict__ WihP, bf16* __restrict__ WhhP,
                        float* __restrict__ biasPre, float* __restrict__ biasHh){
  int gid = blockIdx.x*blockDim.x + threadIdx.x;     // NDIR*G4*(DIN/8) = 262144
  int kc = gid & 63;               // 64 chunks of 8 along K
  int np = (gid >> 6) & (G4-1);    // permuted column n'
  int d  = gid >> 17;              // direction
  int h = np >> 2, g = np & 3;
  int n = g*HID + h;               // original gate row
  int k = kc*8;
  {
    const float* s = Wih + ((size_t)d*G4 + n)*DIN + k;
    bf16* dst = WihP + ((size_t)d*G4 + np)*DIN + k;
    #pragma unroll
    for (int j=0;j<8;j++) dst[j] = (bf16)s[j];
  }
  {
    const float* s = Whh + ((size_t)d*G4 + n)*HID + k;
    bf16* dst = WhhP + ((size_t)d*G4 + np)*HID + k;
    #pragma unroll
    for (int j=0;j<8;j++) dst[j] = (bf16)s[j];
  }
  if (kc == 0){
    biasPre[d*G4 + np] = bih[d*G4 + n];
    biasHh [d*G4 + np] = bhh[d*G4 + n];
  }
}

__global__ void k_init(const float* __restrict__ h0, const float* __restrict__ c0,
                       bf16* __restrict__ hB0, float* __restrict__ cS, float* __restrict__ hS){
  int i = blockIdx.x*blockDim.x + threadIdx.x;   // 2*B*H = 32768
  hB0[i] = (bf16)h0[i];
  cS[i]  = c0[i];
  hS[i]  = h0[i];
}

// ---------------- phase 1: x_pre = x @ W_ih'^T + b_ih (both dirs) ----------------
// C[M=T*B=16384, N=2048], K=512.  Block tile 128x128, K-step 32, 8 waves (2x4).
// Tile staging via global_load_async_to_lds_b128 (ASYNCcnt), no VGPR round trip.
__global__ __launch_bounds__(256)
void k_gemm_xpre(const bf16* __restrict__ xb, const bf16* __restrict__ WihP,
                 const float* __restrict__ biasPre, float* __restrict__ xpre){
  const int d  = blockIdx.z;
  const int m0 = blockIdx.x * 128;
  const int n0 = blockIdx.y * 128;
  const bf16* Wp = WihP + (size_t)d*G4*DIN;
  float* xp = xpre + (size_t)d*T_LEN*BATCH*G4;

  __shared__ __align__(16) bf16 ldsA[128*40];   // pitch 40 bf16 (80B) to dodge bank conflicts
  __shared__ __align__(16) bf16 ldsB[128*40];

  const int tid   = threadIdx.x;
  const int lane  = tid & 31;
  const int wave  = tid >> 5;      // 0..7
  const int wm    = wave & 1;      // M offset 64*wm
  const int wn    = wave >> 1;     // N offset 32*wn
  const int l16   = lane & 15;
  const int lhalf = lane >> 4;     // 0/1

  const unsigned int ldsA_base = (unsigned int)(uintptr_t)&ldsA[0];
  const unsigned int ldsB_base = (unsigned int)(uintptr_t)&ldsB[0];
  const unsigned long long xb_base = (unsigned long long)(uintptr_t)xb;
  const unsigned long long wp_base = (unsigned long long)(uintptr_t)Wp;

  const v8f vzero = {};
  v8f acc[4][2];
  #pragma unroll
  for (int mt=0;mt<4;mt++){ acc[mt][0]=vzero; acc[mt][1]=vzero; }

  for (int k0 = 0; k0 < DIN; k0 += 32){
    #pragma unroll
    for (int c = 0; c < 2; c++){
      int chunk = tid + c*256;     // 512 chunks of 16B per operand tile
      int row = chunk >> 2;
      int kc  = (chunk & 3) * 8;
      unsigned int l_off = (unsigned int)(row*40 + kc) * 2u;
      unsigned int gA = (unsigned int)(((m0+row)*DIN + k0 + kc) * 2);
      unsigned int gB = (unsigned int)(((n0+row)*DIN + k0 + kc) * 2);
      async_load_b128(ldsA_base + l_off, gA, xb_base);
      async_load_b128(ldsB_base + l_off, gB, wp_base);
    }
    if (k0 + 32 < DIN){            // hint next K slab -> global_prefetch_b8
      __builtin_prefetch(xb + (size_t)(m0 + (tid>>1))*DIN + k0 + 32, 0, 0);
      __builtin_prefetch(Wp + (size_t)(n0 + (tid>>1))*DIN + k0 + 32, 0, 0);
    }
    wait_async_all();              // own async writes to LDS done
    __syncthreads();               // everyone's writes visible

    v16bf bfrag[2];
    #pragma unroll
    for (int nt=0; nt<2; nt++){
      const bf16* p = &ldsB[(size_t)(wn*32 + nt*16 + l16)*40 + lhalf*16];
      bfrag[nt] = make_frag(p, p+8);                 // lane: col n=l16, K=(lhalf*16)+e
    }
    #pragma unroll
    for (int mt=0; mt<4; mt++){
      const bf16* p = &ldsA[(size_t)(wm*64 + mt*16 + l16)*40];
      v16bf afrag = make_frag(p + lhalf*16, p + lhalf*16 + 8);  // lane: row m=l16, K=(lhalf*16)+e
      #pragma unroll
      for (int nt=0; nt<2; nt++){
        acc[mt][nt] = __builtin_amdgcn_wmma_f32_16x16x32_bf16(
            false, afrag, false, bfrag[nt], (short)0, acc[mt][nt], false, false);
      }
    }
    __syncthreads();
  }

  #pragma unroll
  for (int mt=0; mt<4; mt++){
    #pragma unroll
    for (int nt=0; nt<2; nt++){
      int col = n0 + wn*32 + nt*16 + l16;
      float bia = biasPre[d*G4 + col];
      #pragma unroll
      for (int i=0;i<8;i++){
        int row = m0 + wm*64 + mt*16 + i + lhalf*8;  // ISA C/D layout
        xp[(size_t)row*G4 + col] = acc[mt][nt][i] + bia;
      }
    }
  }
}

// ---------------- phase 2: one recurrent step (both dirs) ----------------
// Per WG: one dir, 64 permuted gate columns (4 waves x 16).  M=32(batch), K=512.
// h tile (32x512 bf16) staged into LDS by the Tensor Data Mover with row padding
// (pad_interval=7 -> every 1024B, pad_amount=3 -> +16B) giving pitch 520 bf16.
__global__ __launch_bounds__(128)
void k_step(const bf16* __restrict__ WhhP, const float* __restrict__ biasHh,
            const float* __restrict__ xpre, const bf16* __restrict__ hIn,
            bf16* __restrict__ hOut, float* __restrict__ cS, float* __restrict__ hS,
            float* __restrict__ out, int t){
  const int d = blockIdx.y;
  const int t_eff = d ? (T_LEN-1-t) : t;
  const int nblk = blockIdx.x * 64;
  const bf16* Wp = WhhP + (size_t)d*G4*HID;
  const float* xp = xpre + ((size_t)d*T_LEN + t_eff)*BATCH*G4;
  const bf16* hprev = hIn + (size_t)d*BATCH*HID;

  __shared__ __align__(16) bf16 hsh[BATCH*520];       // h tile, pitch 520 bf16 (TDM-padded)
  __shared__ float gsh[4][BATCH][17];                 // per-wave gate tile

  const int tid = threadIdx.x;

  // --- TDM descriptor (D#): 2D tensor, 2B elements, tile 512x32, LDS padding ---
  {
    unsigned long long ga = (unsigned long long)(uintptr_t)hprev;
    u32x4 g0;
    g0[0] = 1u;                                        // count=1 (valid), user mode
    g0[1] = (unsigned int)(uintptr_t)&hsh[0];          // lds_addr (byte offset)
    g0[2] = (unsigned int)ga;                          // global_addr[31:0]
    g0[3] = ((unsigned int)(ga >> 32) & 0x01FFFFFFu)   // global_addr[56:32]
            | 0x80000000u;                             // type=2 ("image")
    u32x8 g1;
    g1[0] = (1u<<16)        // data_size = 1 -> 2 bytes
          | (1u<<20)        // pad_enable
          | (7u<<22)        // pad_interval: 256 DWORDs = 1024B (one 512-bf16 row)
          | (3u<<25);       // pad_amount: 4 DWORDs = 16B  -> pitch 520 bf16
    g1[1] = (512u<<16);     // tensor_dim0 = 512 (bits 79:48)
    g1[2] = (32u<<16);      // tensor_dim1 = 32  (bits 111:80)
    g1[3] = (512u<<16);     // tile_dim0 = 512   (bits 127:112)
    g1[4] = 32u;            // tile_dim1 = 32, tile_dim2 = 0
    g1[5] = 512u;           // tensor_dim0_stride = 512 (bits 207:160 low)
    g1[6] = 0u;             // stride high / dim1_stride low (unused: 2D tile)
    g1[7] = 0u;
    if (tid < 32){          // one wave issues the DMA; TENSORcnt tracks it
      asm volatile("tensor_load_to_lds %0, %1\n\t"
                   "s_wait_tensorcnt 0x0"
                   :: "s"(g0), "s"(g1)
                   : "memory");
    }
  }
  __syncthreads();

  const int lane  = tid & 31;
  const int wave  = tid >> 5;       // 0..3
  const int l16   = lane & 15;
  const int lhalf = lane >> 4;
  const int n0    = nblk + wave*16;

  v8f acc0 = {}, acc1 = {};
  for (int kk=0; kk<HID; kk+=32){
    const bf16* bp = Wp + (size_t)(n0 + l16)*HID + kk + lhalf*16;   // L2-resident W_hh
    if (kk + 32 < HID) __builtin_prefetch(bp + 32, 0, 0);
    v16bf bfrag = make_frag(bp, bp+8);
    const bf16* a0 = &hsh[(size_t)l16*520 + kk];
    const bf16* a1 = &hsh[(size_t)(16 + l16)*520 + kk];
    v16bf af0 = make_frag(a0 + lhalf*16, a0 + lhalf*16 + 8);
    v16bf af1 = make_frag(a1 + lhalf*16, a1 + lhalf*16 + 8);
    acc0 = __builtin_amdgcn_wmma_f32_16x16x32_bf16(false, af0, false, bfrag, (short)0, acc0, false, false);
    acc1 = __builtin_amdgcn_wmma_f32_16x16x32_bf16(false, af1, false, bfrag, (short)0, acc1, false, false);
  }

  // gates = wmma + x_pre + b_hh  -> LDS (cross-lane transpose)
  const int col = n0 + l16;
  const float bia = biasHh[d*G4 + col];
  #pragma unroll
  for (int i=0;i<8;i++){
    int b0 = i + lhalf*8;
    int b1 = 16 + b0;
    gsh[wave][b0][l16] = acc0[i] + xp[(size_t)b0*G4 + col] + bia;
    gsh[wave][b1][l16] = acc1[i] + xp[(size_t)b1*G4 + col] + bia;
  }
  __syncthreads();

  // pointwise LSTM update: lane = batch row, q = local hidden unit (4 per wave)
  const int b = lane;
  #pragma unroll
  for (int q=0;q<4;q++){
    float ig = sigmoidf_(gsh[wave][b][4*q+0]);
    float fg = sigmoidf_(gsh[wave][b][4*q+1]);
    float gg = tanhf    (gsh[wave][b][4*q+2]);
    float og = sigmoidf_(gsh[wave][b][4*q+3]);
    int hIdx = (n0 >> 2) + q;
    size_t si = ((size_t)d*BATCH + b)*HID + hIdx;
    float cn = fg * cS[si] + ig * gg;
    float hn = og * tanhf(cn);
    cS[si] = cn;
    hS[si] = hn;
    hOut[(size_t)d*BATCH*HID + (size_t)b*HID + hIdx] = (bf16)hn;
    out[((size_t)t_eff*BATCH + b)*(2*HID) + (size_t)d*HID + hIdx] = hn;
  }
}

__global__ void k_final(const float* __restrict__ hS, const float* __restrict__ cS,
                        float* __restrict__ out){
  int i = blockIdx.x*blockDim.x + threadIdx.x;   // 2*B*H
  const size_t base = (size_t)T_LEN*BATCH*2*HID;
  out[base + i] = hS[i];
  out[base + 2*BATCH*HID + i] = cS[i];
}

// ---------------- launch ----------------

extern "C" void kernel_launch(void* const* d_in, const int* in_sizes, int n_in,
                              void* d_out, int out_size, void* d_ws, size_t ws_size,
                              hipStream_t stream) {
  (void)in_sizes; (void)n_in; (void)out_size; (void)ws_size;
  const float* x   = (const float*)d_in[0];
  const float* h0  = (const float*)d_in[1];
  const float* c0  = (const float*)d_in[2];
  const float* Wih = (const float*)d_in[3];
  const float* bih = (const float*)d_in[4];
  const float* Whh = (const float*)d_in[5];
  const float* bhh = (const float*)d_in[6];
  float* out = (float*)d_out;

  char* ws = (char*)d_ws;
  size_t off = 0;
  bf16*  xb      = (bf16*)(ws + off);  off += (size_t)T_LEN*BATCH*DIN*2;        // 16 MiB
  bf16*  WihP    = (bf16*)(ws + off);  off += (size_t)NDIR*G4*DIN*2;            // 4 MiB
  bf16*  WhhP    = (bf16*)(ws + off);  off += (size_t)NDIR*G4*HID*2;            // 4 MiB
  float* biasPre = (float*)(ws + off); off += (size_t)NDIR*G4*4;
  float* biasHh  = (float*)(ws + off); off += (size_t)NDIR*G4*4;
  float* xpre    = (float*)(ws + off); off += (size_t)NDIR*T_LEN*BATCH*G4*4;    // 256 MiB
  bf16*  hB      = (bf16*)(ws + off);  off += (size_t)2*NDIR*BATCH*HID*2;       // ping-pong
  float* cS      = (float*)(ws + off); off += (size_t)NDIR*BATCH*HID*4;
  float* hS      = (float*)(ws + off); off += (size_t)NDIR*BATCH*HID*4;

  k_cvt_x<<<(T_LEN*BATCH*DIN)/1024, 256, 0, stream>>>(x, xb);
  k_cvt_w<<<(NDIR*G4*(DIN/8))/256, 256, 0, stream>>>(Wih, bih, Whh, bhh, WihP, WhhP, biasPre, biasHh);
  k_init<<<(NDIR*BATCH*HID)/256, 256, 0, stream>>>(h0, c0, hB, cS, hS);

  k_gemm_xpre<<<dim3((T_LEN*BATCH)/128, G4/128, NDIR), 256, 0, stream>>>(xb, WihP, biasPre, xpre);

  const size_t hStride = (size_t)NDIR*BATCH*HID;
  for (int t = 0; t < T_LEN; ++t){
    const bf16* hIn = hB + (size_t)(t & 1) * hStride;
    bf16* hOut      = hB + (size_t)((t + 1) & 1) * hStride;
    k_step<<<dim3(G4/64, NDIR), 128, 0, stream>>>(WhhP, biasHh, xpre, hIn, hOut, cS, hS, out, t);
  }

  k_final<<<(NDIR*BATCH*HID)/256, 256, 0, stream>>>(hS, cS, out);
}
